// VectorQuantize_83227876262215
// MI455X (gfx1250) — compile-verified, matrix operand order checked by reading, not run on hardware
//
#include <hip/hip_runtime.h>
#include <hip/hip_bf16.h>

// ---------------------------------------------------------------------------
// VQ (VectorQuantize) forward for MI455X / gfx1250, wave32.
//   dist GEMM (137 GFLOP)  -> v_wmma_f32_16x16x32_bf16 (bf16 products, f32 acc)
//   in/out proj (17 GFLOP) -> v_wmma_f32_16x16x4_f32   (exact f32 outputs)
// ---------------------------------------------------------------------------

typedef __attribute__((ext_vector_type(16))) __bf16 v16bf;
typedef __attribute__((ext_vector_type(8)))  __bf16 v8bf;
typedef __attribute__((ext_vector_type(8)))  float  v8f;
typedef __attribute__((ext_vector_type(2)))  float  v2f;

#define DIN   1024
#define DCB   256
#define KCB   8192
#define ROWS  32768   // B*T
#define TPB   4096    // T (rows per batch)

// ---- block-wide sum reduction (LDS + wave32 shuffles) ----------------------
__device__ __forceinline__ float blockReduceSum(float v, float* sm) {
  __syncthreads();                     // protect sm reuse across calls
  const int lane = threadIdx.x & 31;
  const int wid  = threadIdx.x >> 5;
  #pragma unroll
  for (int o = 16; o > 0; o >>= 1) v += __shfl_xor(v, o, 32);
  if (lane == 0) sm[wid] = v;
  __syncthreads();
  if (wid == 0) {
    const int nw = blockDim.x >> 5;
    float r = (lane < nw) ? sm[lane] : 0.0f;
    #pragma unroll
    for (int o = 16; o > 0; o >>= 1) r += __shfl_xor(r, o, 32);
    if (lane == 0) sm[0] = r;
  }
  __syncthreads();
  return sm[0];
}

// ---- weight-norm prep: W_in[i][j] = g[j]*v[i][j]/||v[:,j]||  (1024x256) ----
__global__ void prep_win_kernel(const float* __restrict__ v,
                                const float* __restrict__ g,
                                float* __restrict__ W) {
  __shared__ float sm[32];
  const int j = blockIdx.x;
  float s = 0.0f;
  for (int i = threadIdx.x; i < DIN; i += blockDim.x) {
    const float t = v[(size_t)i * DCB + j];
    s += t * t;
  }
  const float ss = blockReduceSum(s, sm);
  const float gj = g[j] * rsqrtf(ss);
  for (int i = threadIdx.x; i < DIN; i += blockDim.x)
    W[(size_t)i * DCB + j] = v[(size_t)i * DCB + j] * gj;
}

// ---- W_out[i][j] = g[j]*v[i][j]/||v[:,j]||  (256x1024) ---------------------
__global__ void prep_wout_kernel(const float* __restrict__ v,
                                 const float* __restrict__ g,
                                 float* __restrict__ W) {
  __shared__ float sm[32];
  const int j = blockIdx.x;
  const float t = v[(size_t)threadIdx.x * DIN + j];
  const float ss = blockReduceSum(t * t, sm);
  W[(size_t)threadIdx.x * DIN + j] = t * (g[j] * rsqrtf(ss));
}

// ---- cb = normalize(embed_row)*scale + bias  (bf16) ; cbn2 = |cb_bf16|^2 ---
__global__ void prep_cb_kernel(const float* __restrict__ embed,
                               const float* __restrict__ scale,
                               const float* __restrict__ bias,
                               __bf16* __restrict__ cb,
                               float* __restrict__ cbn2) {
  __shared__ float sm[32];
  const int k = blockIdx.x, d = threadIdx.x;
  const float e  = embed[(size_t)k * DCB + d];
  const float ss = blockReduceSum(e * e, sm);
  const float ri = 1.0f / fmaxf(sqrtf(ss), 1e-12f);
  const float c  = e * ri * scale[d] + bias[d];
  const __bf16 cq = (__bf16)c;
  cb[(size_t)k * DCB + d] = cq;
  const float cf = (float)cq;                  // norm of the bf16-rounded row
  const float cn = blockReduceSum(cf * cf, sm);
  if (d == 0) cbn2[k] = cn;
}

// ---- in-projection: z_e = x @ W_in + b_in, f32 WMMA 16x16x4 ----------------
// wave: 16(M) x 64(N); workgroup of 4 waves covers full N=256.
__global__ void gemm_in_kernel(const float* __restrict__ x,
                               const float* __restrict__ W,
                               const float* __restrict__ bias,
                               float* __restrict__ z_e) {
  const int lane  = threadIdx.x & 31;
  const int wave  = threadIdx.x >> 5;       // 0..3
  const int lmod  = lane & 15;
  const int hf    = lane >> 4;              // lane half
  const int rBase = blockIdx.x * 16;
  const int nb    = wave * 64;

  v8f zero = {};
  v8f c[4];
  #pragma unroll
  for (int t = 0; t < 4; ++t) c[t] = zero;

  const float* xrow = x + (size_t)(rBase + lmod) * DIN;
  for (int kc = 0; kc < DIN / 4; ++kc) {
    const int k0 = kc * 4 + hf * 2;                       // A: v0=K0,v1=K1 | halves at K+2
    v2f a = *(const v2f*)(xrow + k0);
    const float* w0 = W + (size_t)k0 * DCB;
    #pragma unroll
    for (int t = 0; t < 4; ++t) {
      const int col = nb + t * 16 + lmod;
      v2f b;
      b[0] = w0[col];
      b[1] = w0[DCB + col];
      c[t] = __builtin_amdgcn_wmma_f32_16x16x4_f32(false, a, false, b,
                                                   (short)0, c[t], false, false);
    }
  }
  #pragma unroll
  for (int t = 0; t < 4; ++t) {
    const int col = nb + t * 16 + lmod;
    const float bv = bias[col];
    #pragma unroll
    for (int j = 0; j < 8; ++j) {
      const int row = rBase + j + hf * 8;                 // C/D: M = vgpr + 8*half
      z_e[(size_t)row * DCB + col] = c[t][j] + bv;
    }
  }
}

// ---- enc = bf16( z_e_row / max(||z_e_row||, eps) ) -------------------------
__global__ void rownorm_kernel(const float* __restrict__ z_e,
                               __bf16* __restrict__ enc) {
  __shared__ float sm[32];
  const int r = blockIdx.x, d = threadIdx.x;
  const float z  = z_e[(size_t)r * DCB + d];
  const float ss = blockReduceSum(z * z, sm);
  const float ri = 1.0f / fmaxf(sqrtf(ss), 1e-12f);
  enc[(size_t)r * DCB + d] = (__bf16)(z * ri);
}

// ---- fused dist + argmin: A = cb tile (codes), B = enc rows ----------------
// score = |cb|^2 - 2*enc.cb  (|enc|^2 is constant per row -> dropped).
// Each wave owns 16 enc rows, keeps all 8 B-fragments resident, streams 512
// code tiles of 16 through v_wmma_f32_16x16x32_bf16 (8 wmma per tile).
__global__ void dist_argmin_kernel(const __bf16* __restrict__ enc,
                                   const __bf16* __restrict__ cb,
                                   const float* __restrict__ cbn2,
                                   int* __restrict__ idxOut) {
  const int lane  = threadIdx.x & 31;
  const int wave  = threadIdx.x >> 5;          // 0..7
  const int lmod  = lane & 15;
  const int hf    = lane >> 4;
  const int rBase = blockIdx.x * 128 + wave * 16;

  // resident B fragments: 16-bit B 32x16 layout (mirror of 16-bit A 16x32):
  // lane half 0 -> K {0..7,16..23}, half 1 -> K {8..15,24..31}, N = lane&15
  v16bf bfrag[8];
  const __bf16* encRow = enc + (size_t)(rBase + lmod) * DCB;
  #pragma unroll
  for (int cI = 0; cI < 8; ++cI) {
    const int k0 = cI * 32;
    v8bf lo = *(const v8bf*)(encRow + k0 + hf * 8);
    v8bf hi = *(const v8bf*)(encRow + k0 + 16 + hf * 8);
    bfrag[cI] = __builtin_shufflevector(lo, hi, 0, 1, 2, 3, 4, 5, 6, 7,
                                        8, 9, 10, 11, 12, 13, 14, 15);
  }

  float best = 3.4e38f;
  int   bidx = 0;
  for (int t = 0; t < 512; ++t) {
    const int m0 = t * 16;
    const __bf16* cbRow = cb + (size_t)(m0 + lmod) * DCB;
    __builtin_prefetch(cbRow + 16 * DCB, 0, 1);           // next code tile
    v8f acc = {};
    #pragma unroll
    for (int cI = 0; cI < 8; ++cI) {
      const int k0 = cI * 32;
      v8bf lo = *(const v8bf*)(cbRow + k0 + hf * 8);
      v8bf hi = *(const v8bf*)(cbRow + k0 + 16 + hf * 8);
      v16bf af = __builtin_shufflevector(lo, hi, 0, 1, 2, 3, 4, 5, 6, 7,
                                         8, 9, 10, 11, 12, 13, 14, 15);
      acc = __builtin_amdgcn_wmma_f32_16x16x32_bf16(false, af, false, bfrag[cI],
                                                    (short)0, acc, false, false);
    }
    // local argmin over the 8 codes this lane holds (codes m0 + hf*8 + j)
    v8f n2 = *(const v8f*)(cbn2 + m0 + hf * 8);
    float lb = 3.4e38f;
    int   li = 0;
    #pragma unroll
    for (int j = 0; j < 8; ++j) {
      const float s = n2[j] - 2.0f * acc[j];
      const int code = m0 + hf * 8 + j;
      if (s < lb) { lb = s; li = code; }
    }
    // merge the two lane halves (same enc row, other 8 codes)
    const float ob = __shfl_xor(lb, 16, 32);
    const int   oi = __shfl_xor(li, 16, 32);
    if (ob < lb || (ob == lb && oi < li)) { lb = ob; li = oi; }
    if (lb < best || (lb == best && li < bidx)) { best = lb; bidx = li; }
  }
  if (lane < 16) idxOut[rBase + lmod] = bidx;
}

// ---- losses (commitment == codebook fwd value), usage histogram, indices ---
__global__ void loss_usage_kernel(const int* __restrict__ idx,
                                  const float* __restrict__ embed,
                                  const float* __restrict__ z_e,
                                  float* __restrict__ lossSum,
                                  float* __restrict__ usage,
                                  float* __restrict__ indices) {
  __shared__ float sm[32];
  const int r = blockIdx.x, d = threadIdx.x;
  const int k = idx[r];
  const float diff = z_e[(size_t)r * DCB + d] - embed[(size_t)k * DCB + d];
  const float ss = blockReduceSum(diff * diff, sm);
  if (d == 0) {
    atomicAdd(&lossSum[r / TPB], ss);
    atomicAdd(&usage[k], 1.0f);
    indices[r] = (float)k;        // d_out is f32; indices emitted as values
  }
}

// ---- out-projection: out = embed[idx] @ W_out + b_out, f32 WMMA ------------
// A-fragment rows are gathered straight from embed via idx (no z_q buffer).
__global__ void gemm_out_kernel(const int* __restrict__ idx,
                                const float* __restrict__ embed,
                                const float* __restrict__ W,
                                const float* __restrict__ bias,
                                float* __restrict__ out) {
  const int lane  = threadIdx.x & 31;
  const int wave  = threadIdx.x >> 5;
  const int lmod  = lane & 15;
  const int hf    = lane >> 4;
  const int rBase = blockIdx.x * 16;
  const int nb    = blockIdx.y * 256 + wave * 64;

  const int code = idx[rBase + lmod];
  const float* arow = embed + (size_t)code * DCB;

  v8f zero = {};
  v8f c[4];
  #pragma unroll
  for (int t = 0; t < 4; ++t) c[t] = zero;

  for (int kc = 0; kc < DCB / 4; ++kc) {
    const int k0 = kc * 4 + hf * 2;
    v2f a = *(const v2f*)(arow + k0);
    const float* w0 = W + (size_t)k0 * DIN;
    #pragma unroll
    for (int t = 0; t < 4; ++t) {
      const int col = nb + t * 16 + lmod;
      v2f b;
      b[0] = w0[col];
      b[1] = w0[DIN + col];
      c[t] = __builtin_amdgcn_wmma_f32_16x16x4_f32(false, a, false, b,
                                                   (short)0, c[t], false, false);
    }
  }
  #pragma unroll
  for (int t = 0; t < 4; ++t) {
    const int col = nb + t * 16 + lmod;
    const float bv = bias[col];
    #pragma unroll
    for (int j = 0; j < 8; ++j) {
      const int row = rBase + j + hf * 8;
      out[(size_t)row * DIN + col] = c[t][j] + bv;
    }
  }
}

// ---- zero accumulators (runs every launch: deterministic) ------------------
__global__ void init_kernel(float* __restrict__ lossSum, float* __restrict__ usage) {
  const int i = blockIdx.x * blockDim.x + threadIdx.x;
  if (i < KCB) usage[i] = 0.0f;
  if (i < 8)   lossSum[i] = 0.0f;
}

__global__ void finalize_kernel(const float* __restrict__ lossSum,
                                float* __restrict__ commit,
                                float* __restrict__ codebk) {
  const int b = threadIdx.x;
  if (b < 8) {
    const float v = lossSum[b] * (1.0f / (float)(TPB * DCB));
    commit[b] = v;
    codebk[b] = v;
  }
}

extern "C" void kernel_launch(void* const* d_in, const int* in_sizes, int n_in,
                              void* d_out, int out_size, void* d_ws, size_t ws_size,
                              hipStream_t stream) {
  const float* x          = (const float*)d_in[0];
  const float* embed      = (const float*)d_in[1];
  const float* v_in       = (const float*)d_in[2];
  const float* g_in       = (const float*)d_in[3];
  const float* b_in       = (const float*)d_in[4];
  const float* v_out      = (const float*)d_in[5];
  const float* g_out      = (const float*)d_in[6];
  const float* b_out      = (const float*)d_in[7];
  const float* code_scale = (const float*)d_in[8];
  const float* code_bias  = (const float*)d_in[9];

  // workspace layout (bytes)
  char* ws = (char*)d_ws;
  float*  W_in    = (float*)(ws + 0);              //  1 MB (1024x256 f32)
  float*  W_out   = (float*)(ws + 1048576);        //  1 MB (256x1024 f32)
  __bf16* cb      = (__bf16*)(ws + 2097152);       //  4 MB (8192x256 bf16)
  float*  cbn2    = (float*)(ws + 6291456);        // 32 KB
  __bf16* enc     = (__bf16*)(ws + 6324224);       // 16 MB (32768x256 bf16)
  int*    idxb    = (int*)(ws + 23101440);         // 128 KB
  float*  lossSum = (float*)(ws + 23232512);       // 32 B

  // d_out layout (f32 elements), reference tuple order:
  // out, commitment_loss, codebook_loss, indices, z_e, usage
  float* out     = (float*)d_out;
  float* commit  = out + (size_t)ROWS * DIN;       // 33554432
  float* codebk  = commit + 8;
  float* indices = codebk + 8;
  float* z_e     = indices + ROWS;
  float* usage   = z_e + (size_t)ROWS * DCB;

  prep_win_kernel  <<<DCB,  256, 0, stream>>>(v_in, g_in, W_in);
  prep_wout_kernel <<<DIN,  256, 0, stream>>>(v_out, g_out, W_out);
  prep_cb_kernel   <<<KCB,  256, 0, stream>>>(embed, code_scale, code_bias, cb, cbn2);
  init_kernel      <<<32,   256, 0, stream>>>(lossSum, usage);

  gemm_in_kernel   <<<ROWS / 16, 128, 0, stream>>>(x, W_in, b_in, z_e);
  rownorm_kernel   <<<ROWS, 256, 0, stream>>>(z_e, enc);
  dist_argmin_kernel<<<ROWS / 128, 256, 0, stream>>>(enc, cb, cbn2, idxb);
  loss_usage_kernel<<<ROWS, 256, 0, stream>>>(idxb, embed, z_e, lossSum, usage, indices);
  gemm_out_kernel  <<<dim3(ROWS / 16, 4), 128, 0, stream>>>(idxb, embed, W_out, b_out, out);
  finalize_kernel  <<<1, 8, 0, stream>>>(lossSum, commit, codebk);
}
